// Encoder_two_fc_17540646437762
// MI455X (gfx1250) — matrix operand order repaired and checked by reading, not compiled, not run on hardware
//
#include <hip/hip_runtime.h>
#include <hip/hip_bf16.h>
#include <math.h>

// ---------------------------------------------------------------------------
// MI455X (gfx1250) two-stream LSTM encoder.
//   - bf16 WMMA (v_wmma_f32_16x16x32_bf16) with fp32 accumulation everywhere.
//   - Weights + activations pre-converted to bf16 once; GEMM staging is a pure
//     copy via GLOBAL_LOAD_ASYNC_TO_LDS_B128 (ASYNCcnt-tracked), 3-stage
//     rotating LDS pipeline: s_wait_asynccnt 4 -> barrier -> prefetch it+2 ->
//     wmma tile it. Per-thread global base pointers hoisted out of the K loop.
//   - fp32 kept only where numerics demand it: pre-BN Z, cell state c, gate
//     pre-activations, final output.
// Dims: B=128, T=256, D=512, F0=2048, F1=1024.
// ---------------------------------------------------------------------------

#define Bv   128
#define Tv   256
#define Dv   512
#define F0v  2048
#define F1v  1024
#define EPSv 1e-5f

typedef __attribute__((ext_vector_type(8)))  __bf16 v8bf;
typedef __attribute__((ext_vector_type(16))) __bf16 v16bf;
typedef __attribute__((ext_vector_type(8)))  float  v8f;

#define BM 128
#define BN 128
#define BK 32
#define NSTAGE 3

__device__ __forceinline__ unsigned short f2bf(float f) {
    union { float f; unsigned int u; } v; v.f = f;
    unsigned int u = v.u;
    unsigned int r = (u + 0x7FFFu + ((u >> 16) & 1u)) >> 16;   // RNE
    return (unsigned short)r;
}
__device__ __forceinline__ float bf2f(unsigned short b) {
    union { unsigned int u; float f; } v; v.u = ((unsigned int)b) << 16;
    return v.f;
}
__device__ __forceinline__ float sigmoidf_(float x) {
    return 1.0f / (1.0f + __expf(-x));
}

// 16-byte per-lane async copy global -> LDS (GLOBAL_LOAD_ASYNC_TO_LDS_B128).
__device__ __forceinline__ void async_cp16(unsigned lds_off, const void* gaddr) {
    asm volatile("global_load_async_to_lds_b128 %0, %1, off"
                 :: "v"(lds_off), "v"(gaddr) : "memory");
}
template<int N>
__device__ __forceinline__ void wait_async() {
    asm volatile("s_wait_asynccnt %0" :: "n"(N) : "memory");
}

// Shared WMMA tile compute: 8x v_wmma per 128x128x32 tile, 8 waves (2M x 4N).
__device__ __forceinline__ void wmma_tile(const unsigned short (*As)[BK],
                                          const unsigned short (*Ws)[BK],
                                          int lane, int waveM, int waveN,
                                          v8f acc[4][2]) {
    const int lr      = lane & 15;
    const int a_khalf = (lane >> 4) * 16;   // A frag: 16 contiguous K of A row
    const int b_koff  = (lane >> 4) * 16;   // B frag: 16 contiguous K of W row

    v16bf bfr[2];
    #pragma unroll
    for (int nt = 0; nt < 2; ++nt)
        bfr[nt] = *(const v16bf*)&Ws[waveN * 32 + nt * 16 + lr][b_koff];

    #pragma unroll
    for (int mt = 0; mt < 4; ++mt) {
        const unsigned short* pa = &As[waveM * 64 + mt * 16 + lr][0];
        v8bf lo = *(const v8bf*)(pa + a_khalf);
        v8bf hi = *(const v8bf*)(pa + 8 + a_khalf);
        v16bf afr;
        #pragma unroll
        for (int i = 0; i < 8; ++i) { afr[i] = lo[i]; afr[i + 8] = hi[i]; }
        #pragma unroll
        for (int nt = 0; nt < 2; ++nt)
            acc[mt][nt] = __builtin_amdgcn_wmma_f32_16x16x32_bf16(
                false, afr, false, bfr[nt], (short)0, acc[mt][nt], false, false);
    }
}

// ---------------------------------------------------------------------------
// bf16 dual-operand GEMM, async-LDS 3-stage pipeline.
// C[m,n] = epi( A0@W0^T + A1@W1^T + bias0[n] + bias1[n] ), all bf16 in,
// fp32 out. Dims multiples of tiles. epilogue: 0=none, 1=tanh.
// ---------------------------------------------------------------------------
__launch_bounds__(256)
__global__ void wmma_gemm_bf16_kernel(
    const unsigned short* __restrict__ A0, long a0_rs,
    const unsigned short* __restrict__ W0, int K0,
    const unsigned short* __restrict__ A1, long a1_rs,
    const unsigned short* __restrict__ W1, int K1,
    const float* __restrict__ bias0, const float* __restrict__ bias1,
    float* __restrict__ C, long c_rs, int epilogue)
{
    __shared__ alignas(32) unsigned short As[NSTAGE][BM][BK];
    __shared__ alignas(32) unsigned short Ws[NSTAGE][BN][BK];

    const int tid   = threadIdx.x;
    const int lane  = tid & 31;
    const int wave  = tid >> 5;
    const int waveM = wave & 1;
    const int waveN = wave >> 1;
    const int mblk  = blockIdx.y * BM;
    const int nblk  = blockIdx.x * BN;

    const int srow = tid >> 1;          // staging: half row (16 bf16 = 32B)
    const int scol = (tid & 1) * 16;

    const int n0 = K0 / BK;             // tiles in operand 0
    const int NT = (K0 + K1) / BK;      // total K tiles

    // per-thread global base pointers, hoisted out of the K loop
    const unsigned short* ga0 = A0 + (size_t)(mblk + srow) * (size_t)a0_rs + scol;
    const unsigned short* gw0 = W0 + (size_t)(nblk + srow) * (size_t)K0 + scol;
    const unsigned short* ga1 = A1 ? A1 + (size_t)(mblk + srow) * (size_t)a1_rs + scol : nullptr;
    const unsigned short* gw1 = A1 ? W1 + (size_t)(nblk + srow) * (size_t)K1 + scol : nullptr;

    v8f acc[4][2] = {};

    // issue 4 async b128 copies for K-tile `it` into LDS stage buffer `buf`
    auto stage = [&](int buf, int it) {
        const unsigned short* ga;
        const unsigned short* gw;
        if (it < n0) { ga = ga0 + (size_t)it * BK;        gw = gw0 + (size_t)it * BK; }
        else         { ga = ga1 + (size_t)(it - n0) * BK; gw = gw1 + (size_t)(it - n0) * BK; }
        unsigned la = (unsigned)(size_t)&As[buf][srow][scol];
        unsigned lw = (unsigned)(size_t)&Ws[buf][srow][scol];
        async_cp16(la,      ga);
        async_cp16(la + 16, ga + 8);
        async_cp16(lw,      gw);
        async_cp16(lw + 16, gw + 8);
    };

    stage(0, 0);
    if (NT > 1) stage(1, 1);

    int buf = 0;
    for (int it = 0; it < NT; ++it) {
        // tile `it` is oldest outstanding; tile it+1 (4 ops) may stay in flight
        if (it + 1 < NT) wait_async<4>(); else wait_async<0>();
        __syncthreads();   // all waves: tile `it` resident; tile it-1 consumed
        if (it + 2 < NT) stage((it + 2) % NSTAGE, it + 2);
        wmma_tile(As[buf], Ws[buf], lane, waveM, waveN, acc);
        buf = (buf + 1) % NSTAGE;
    }

    const int nlane = lane & 15;
    const int moff  = (lane >> 4) * 8;
    #pragma unroll
    for (int mt = 0; mt < 4; ++mt) {
        #pragma unroll
        for (int nt = 0; nt < 2; ++nt) {
            const int n = nblk + waveN * 32 + nt * 16 + nlane;
            float bsum = (bias0 ? bias0[n] : 0.0f) + (bias1 ? bias1[n] : 0.0f);
            #pragma unroll
            for (int r = 0; r < 8; ++r) {
                const int m = mblk + waveM * 64 + mt * 16 + moff + r;
                float v = acc[mt][nt][r] + bsum;
                if (epilogue == 1) v = tanhf(v);
                C[(size_t)m * (size_t)c_rs + n] = v;
            }
        }
    }
}

// ---------------------------------------------------------------------------
// Embedding GEMM: A fp32 (feat), W bf16 (pre-converted), fp32 out (pre-BN Z).
// Runs once per stream; converts A in staging.
// ---------------------------------------------------------------------------
__launch_bounds__(256)
__global__ void wmma_gemm_f32a_kernel(
    const float* __restrict__ A, long a_rs,
    const unsigned short* __restrict__ W, int K,
    const float* __restrict__ bias,
    float* __restrict__ C, long c_rs)
{
    __shared__ alignas(32) unsigned short As[BM][BK];
    __shared__ alignas(32) unsigned short Ws[BN][BK];

    const int tid   = threadIdx.x;
    const int lane  = tid & 31;
    const int wave  = tid >> 5;
    const int waveM = wave & 1;
    const int waveN = wave >> 1;
    const int mblk  = blockIdx.y * BM;
    const int nblk  = blockIdx.x * BN;
    const int srow  = tid >> 1;
    const int scol  = (tid & 1) * 16;

    const float* srcA          = A + (size_t)(mblk + srow) * (size_t)a_rs + scol;
    const unsigned short* srcW = W + (size_t)(nblk + srow) * (size_t)K + scol;

    v8f acc[4][2] = {};

    for (int kb = 0; kb < K; kb += BK) {
        __syncthreads();
        {   // A: fp32 -> bf16
            const float* src = srcA + kb;
            unsigned short tmp[16];
            #pragma unroll
            for (int i = 0; i < 16; ++i) tmp[i] = f2bf(src[i]);
            *(uint4*)&As[srow][scol]     = *(const uint4*)&tmp[0];
            *(uint4*)&As[srow][scol + 8] = *(const uint4*)&tmp[8];
        }
        {   // W: bf16 copy
            const unsigned short* src = srcW + kb;
            *(uint4*)&Ws[srow][scol]     = *(const uint4*)&src[0];
            *(uint4*)&Ws[srow][scol + 8] = *(const uint4*)&src[8];
        }
        __syncthreads();
        wmma_tile(As, Ws, lane, waveM, waveN, acc);
    }

    const int nlane = lane & 15;
    const int moff  = (lane >> 4) * 8;
    #pragma unroll
    for (int mt = 0; mt < 4; ++mt) {
        #pragma unroll
        for (int nt = 0; nt < 2; ++nt) {
            const int n = nblk + waveN * 32 + nt * 16 + nlane;
            float b = bias ? bias[n] : 0.0f;
            #pragma unroll
            for (int r = 0; r < 8; ++r) {
                const int m = mblk + waveM * 64 + mt * 16 + moff + r;
                C[(size_t)m * (size_t)c_rs + n] = acc[mt][nt][r] + b;
            }
        }
    }
}

// ---- fp32 -> bf16 conversion (weights, one-shot) ------------------------
__global__ void cvt_bf16_kernel(const float* __restrict__ src,
                                unsigned short* __restrict__ dst, size_t n)
{
    size_t i = (size_t)blockIdx.x * blockDim.x + threadIdx.x;
    const size_t stride = (size_t)gridDim.x * blockDim.x;
    for (; i < n; i += stride) dst[i] = f2bf(src[i]);
}

// ---- BatchNorm stats over columns (coalesced atomics) -------------------
__global__ void bn_colstats_kernel(const float* __restrict__ Z,
                                   float* __restrict__ sum,
                                   float* __restrict__ sumsq)
{
    const int r0 = blockIdx.x * 128;
    const int c  = threadIdx.x;              // cols c and c+256
    float s0 = 0.f, q0 = 0.f, s1 = 0.f, q1 = 0.f;
    for (int r = 0; r < 128; ++r) {
        const float* row = Z + (size_t)(r0 + r) * Dv;
        float a = row[c];       s0 += a; q0 += a * a;
        float b = row[c + 256]; s1 += b; q1 += b * b;
    }
    atomicAdd(&sum[c], s0);        atomicAdd(&sumsq[c], q0);
    atomicAdd(&sum[c + 256], s1);  atomicAdd(&sumsq[c + 256], q1);
}

__global__ void bn_finalize_kernel(const float* __restrict__ sum,
                                   const float* __restrict__ sumsq,
                                   const float* __restrict__ gamma,
                                   const float* __restrict__ beta,
                                   float* __restrict__ scale,
                                   float* __restrict__ shift,
                                   float inv_rows)
{
    const int c = threadIdx.x;               // 512 threads
    float mu  = sum[c] * inv_rows;
    float var = sumsq[c] * inv_rows - mu * mu;
    float sc  = gamma[c] * rsqrtf(var + EPSv);
    scale[c] = sc;
    shift[c] = beta[c] - mu * sc;
}

// ---- BN + ReLU, fp32 Z -> bf16 e ----------------------------------------
__global__ void bn_apply_relu_kernel(const float* __restrict__ Z,
                                     unsigned short* __restrict__ e,
                                     const float* __restrict__ scale,
                                     const float* __restrict__ shift,
                                     size_t n)
{
    size_t i = (size_t)blockIdx.x * blockDim.x + threadIdx.x;
    const size_t stride = (size_t)gridDim.x * blockDim.x;
    for (; i < n; i += stride) {
        int c = (int)(i & (Dv - 1));
        float v = Z[i] * scale[c] + shift[c];
        e[i] = f2bf(v > 0.f ? v : 0.f);
    }
}

// ---- LSTM pointwise for both cells: c fp32 in/out, h out bf16 -----------
__global__ void lstm_pointwise_kernel(const float* __restrict__ gates0,
                                      const float* __restrict__ gates1,
                                      unsigned short* __restrict__ h0, float* __restrict__ c0,
                                      unsigned short* __restrict__ h1, float* __restrict__ c1,
                                      const float* __restrict__ mask, int t)
{
    const int idx  = blockIdx.x * blockDim.x + threadIdx.x;  // 0 .. 2*B*D-1
    const int cell = idx >> 16;                              // B*D = 65536
    const int r    = idx & 65535;
    const int b    = r >> 9;
    const int d    = r & (Dv - 1);
    const float* g = (cell ? gates1 : gates0) + (size_t)b * 4 * Dv;
    unsigned short* hp = cell ? h1 : h0;
    float*          cp = cell ? c1 : c0;

    float ig = sigmoidf_(g[d]);
    float fg = sigmoidf_(g[Dv + d]);
    float gg = tanhf(g[2 * Dv + d]);
    float og = sigmoidf_(g[3 * Dv + d]);
    float m  = mask[(size_t)b * Tv + t];

    float cv = fg * cp[r] + ig * gg;
    float hv = og * tanhf(cv);
    cp[r] = cv * m;
    hp[r] = f2bf(hv * m);
}

// ---- cross gating: o0[:,t]=sig(s0)*h0, o1[:,t]=sig(s1)*h1 (bf16 out) ----
__global__ void crossgate_kernel(const float* __restrict__ s0,
                                 const float* __restrict__ s1,
                                 const unsigned short* __restrict__ h0,
                                 const unsigned short* __restrict__ h1,
                                 unsigned short* __restrict__ o0,
                                 unsigned short* __restrict__ o1, int t)
{
    const int idx = blockIdx.x * blockDim.x + threadIdx.x;   // 0 .. B*D-1
    const int b = idx >> 9;
    const int d = idx & (Dv - 1);
    const size_t ofs = ((size_t)b * Tv + t) * Dv + d;
    o0[ofs] = f2bf(sigmoidf_(s0[idx]) * bf2f(h0[idx]));
    o1[ofs] = f2bf(sigmoidf_(s1[idx]) * bf2f(h1[idx]));
}

extern "C" void kernel_launch(void* const* d_in, const int* in_sizes, int n_in,
                              void* d_out, int out_size, void* d_ws, size_t ws_size,
                              hipStream_t stream) {
    (void)in_sizes; (void)n_in; (void)out_size; (void)ws_size;

    const float* feat0  = (const float*)d_in[0];
    const float* feat1  = (const float*)d_in[1];
    const float* mask   = (const float*)d_in[2];
    const float* w_emb0 = (const float*)d_in[3];
    const float* b_emb0 = (const float*)d_in[4];
    const float* gamma0 = (const float*)d_in[5];
    const float* beta0  = (const float*)d_in[6];
    const float* w_emb1 = (const float*)d_in[7];
    const float* b_emb1 = (const float*)d_in[8];
    const float* gamma1 = (const float*)d_in[9];
    const float* beta1  = (const float*)d_in[10];
    const float* w_ih0  = (const float*)d_in[11];
    const float* w_hh0  = (const float*)d_in[12];
    const float* b_ih0  = (const float*)d_in[13];
    const float* b_hh0  = (const float*)d_in[14];
    const float* w_ih1  = (const float*)d_in[15];
    const float* w_hh1  = (const float*)d_in[16];
    const float* b_ih1  = (const float*)d_in[17];
    const float* b_hh1  = (const float*)d_in[18];
    const float* wg0    = (const float*)d_in[19];
    const float* bg0    = (const float*)d_in[20];
    const float* wg1    = (const float*)d_in[21];
    const float* bg1    = (const float*)d_in[22];
    const float* wf1    = (const float*)d_in[23];
    const float* wf2    = (const float*)d_in[24];
    const float* bf     = (const float*)d_in[25];
    float* out = (float*)d_out;

    // ---- workspace carve (256B-aligned regions) ----
    const size_t BT = (size_t)Bv * Tv;           // 32768
    char* base = (char*)d_ws;
    size_t off = 0;
    auto carve = [&](size_t bytes) {
        char* p = base + off;
        off += (bytes + 255) & ~(size_t)255;
        return (void*)p;
    };
    // bf16 weights
    unsigned short* we0  = (unsigned short*)carve((size_t)Dv * F0v * 2);
    unsigned short* we1  = (unsigned short*)carve((size_t)Dv * F1v * 2);
    unsigned short* wih0 = (unsigned short*)carve((size_t)4 * Dv * Dv * 2);
    unsigned short* whh0 = (unsigned short*)carve((size_t)4 * Dv * Dv * 2);
    unsigned short* wih1 = (unsigned short*)carve((size_t)4 * Dv * Dv * 2);
    unsigned short* whh1 = (unsigned short*)carve((size_t)4 * Dv * Dv * 2);
    unsigned short* wgg0 = (unsigned short*)carve((size_t)Dv * Dv * 2);
    unsigned short* wgg1 = (unsigned short*)carve((size_t)Dv * Dv * 2);
    unsigned short* wff1 = (unsigned short*)carve((size_t)Dv * Dv * 2);
    unsigned short* wff2 = (unsigned short*)carve((size_t)Dv * Dv * 2);
    // bf16 activations
    unsigned short* e0bf = (unsigned short*)carve(BT * Dv * 2);
    unsigned short* e1bf = (unsigned short*)carve(BT * Dv * 2);
    unsigned short* o0bf = (unsigned short*)carve(BT * Dv * 2);
    unsigned short* o1bf = (unsigned short*)carve(BT * Dv * 2);
    unsigned short* hbf  = (unsigned short*)carve((size_t)2 * Bv * Dv * 2); // h0,h1
    unsigned short* h0bf = hbf, *h1bf = hbf + (size_t)Bv * Dv;
    // fp32 scratch
    float* Z      = (float*)carve(BT * Dv * 4);          // pre-BN, reused per stream
    float* gates0 = (float*)carve((size_t)Bv * 4 * Dv * 4);
    float* gates1 = (float*)carve((size_t)Bv * 4 * Dv * 4);
    float* s0     = (float*)carve((size_t)Bv * Dv * 4);
    float* s1     = (float*)carve((size_t)Bv * Dv * 4);
    float* cc     = (float*)carve((size_t)2 * Bv * Dv * 4);  // c0,c1
    float* c0 = cc, *c1 = cc + (size_t)Bv * Dv;
    float* stats  = (float*)carve((size_t)4 * Dv * 4);   // sum,sq,scale,shift
    float* bsum = stats, *bsq = stats + Dv, *bsc = stats + 2 * Dv, *bsh = stats + 3 * Dv;

    const dim3 blk(256);
    const long ldD = Dv, ldTD = (long)Tv * Dv, ld4D = 4 * Dv;

    // ---- zero init: h (bf16 zeros == 0 bits), c ----
    hipMemsetAsync(hbf, 0, (size_t)2 * Bv * Dv * 2, stream);
    hipMemsetAsync(cc,  0, (size_t)2 * Bv * Dv * 4, stream);

    // ---- one-shot weight conversion fp32 -> bf16 ----
    auto cvt = [&](const float* s, unsigned short* d, size_t n) {
        cvt_bf16_kernel<<<dim3(512), blk, 0, stream>>>(s, d, n);
    };
    cvt(w_emb0, we0, (size_t)Dv * F0v);
    cvt(w_emb1, we1, (size_t)Dv * F1v);
    cvt(w_ih0, wih0, (size_t)4 * Dv * Dv);
    cvt(w_hh0, whh0, (size_t)4 * Dv * Dv);
    cvt(w_ih1, wih1, (size_t)4 * Dv * Dv);
    cvt(w_hh1, whh1, (size_t)4 * Dv * Dv);
    cvt(wg0, wgg0, (size_t)Dv * Dv);
    cvt(wg1, wgg1, (size_t)Dv * Dv);
    cvt(wf1, wff1, (size_t)Dv * Dv);
    cvt(wf2, wff2, (size_t)Dv * Dv);

    const float inv_rows = 1.0f / (float)BT;

    // ---- stream 0: embed -> BN stats -> BN+ReLU (bf16) ----
    hipMemsetAsync(stats, 0, 2 * (size_t)Dv * 4, stream);
    wmma_gemm_f32a_kernel<<<dim3(Dv / BN, BT / BM), blk, 0, stream>>>(
        feat0, (long)F0v, we0, F0v, b_emb0, Z, ldD);
    bn_colstats_kernel<<<dim3(BT / 128), blk, 0, stream>>>(Z, bsum, bsq);
    bn_finalize_kernel<<<dim3(1), dim3(Dv), 0, stream>>>(bsum, bsq, gamma0, beta0, bsc, bsh, inv_rows);
    bn_apply_relu_kernel<<<dim3(4096), blk, 0, stream>>>(Z, e0bf, bsc, bsh, BT * Dv);

    // ---- stream 1 (reuses Z and stats) ----
    hipMemsetAsync(stats, 0, 2 * (size_t)Dv * 4, stream);
    wmma_gemm_f32a_kernel<<<dim3(Dv / BN, BT / BM), blk, 0, stream>>>(
        feat1, (long)F1v, we1, F1v, b_emb1, Z, ldD);
    bn_colstats_kernel<<<dim3(BT / 128), blk, 0, stream>>>(Z, bsum, bsq);
    bn_finalize_kernel<<<dim3(1), dim3(Dv), 0, stream>>>(bsum, bsq, gamma1, beta1, bsc, bsh, inv_rows);
    bn_apply_relu_kernel<<<dim3(4096), blk, 0, stream>>>(Z, e1bf, bsc, bsh, BT * Dv);

    // ---- recurrence over T timesteps ----
    for (int t = 0; t < Tv; ++t) {
        // gates = x_t @ w_ih^T + h @ w_hh^T + (b_ih + b_hh), per cell
        wmma_gemm_bf16_kernel<<<dim3(4 * Dv / BN, Bv / BM), blk, 0, stream>>>(
            e0bf + (size_t)t * Dv, ldTD, wih0, Dv,
            h0bf, ldD, whh0, Dv,
            b_ih0, b_hh0, gates0, ld4D, 0);
        wmma_gemm_bf16_kernel<<<dim3(4 * Dv / BN, Bv / BM), blk, 0, stream>>>(
            e1bf + (size_t)t * Dv, ldTD, wih1, Dv,
            h1bf, ldD, whh1, Dv,
            b_ih1, b_hh1, gates1, ld4D, 0);

        // LSTM cell updates + mask (both cells); h written as bf16
        lstm_pointwise_kernel<<<dim3(2 * Bv * Dv / 256), blk, 0, stream>>>(
            gates0, gates1, h0bf, c0, h1bf, c1, mask, t);

        // cross-gate projections: s0 = h1@wg0^T+bg0 ; s1 = h0@wg1^T+bg1
        wmma_gemm_bf16_kernel<<<dim3(Dv / BN, Bv / BM), blk, 0, stream>>>(
            h1bf, ldD, wgg0, Dv, (const unsigned short*)nullptr, 0, (const unsigned short*)nullptr, 0,
            bg0, nullptr, s0, ldD, 0);
        wmma_gemm_bf16_kernel<<<dim3(Dv / BN, Bv / BM), blk, 0, stream>>>(
            h0bf, ldD, wgg1, Dv, (const unsigned short*)nullptr, 0, (const unsigned short*)nullptr, 0,
            bg1, nullptr, s1, ldD, 0);

        crossgate_kernel<<<dim3(Bv * Dv / 256), blk, 0, stream>>>(
            s0, s1, h0bf, h1bf, o0bf, o1bf, t);
    }

    // ---- final: out = tanh(o0@wf1^T + o1@wf2^T + bf) ----
    wmma_gemm_bf16_kernel<<<dim3(Dv / BN, BT / BM), blk, 0, stream>>>(
        o0bf, ldD, wff1, Dv, o1bf, ldD, wff2, Dv,
        bf, nullptr, out, ldD, 1);
}
